// LinformerAttention_7499012899641
// MI455X (gfx1250) — compile-verified, matrix operand order checked by reading, not run on hardware
//
#include <hip/hip_runtime.h>
#include <hip/hip_bf16.h>

// ---------------------------------------------------------------------------
// Linformer attention for MI455X (gfx1250, wave32, WMMA).
// - All large matmuls through v_wmma_f32_16x16x32_bf16 (f32 accumulate).
// - Flash-fused attention: 134M-element score tensor never materialized.
// - LDS staging via GLOBAL_LOAD_ASYNC_TO_LDS_B128 + s_wait_asynccnt with
//   double-buffered K-loop (falls back to sync staging if builtins absent).
// ---------------------------------------------------------------------------

typedef __attribute__((ext_vector_type(16))) __bf16 v16bf;
typedef __attribute__((ext_vector_type(8)))  float  v8f;
typedef __attribute__((ext_vector_type(4)))  int    v4i;

#define WMMA_BF16(a, b, c) \
  __builtin_amdgcn_wmma_f32_16x16x32_bf16(false, (a), false, (b), (short)0, (c), false, false)

#if defined(__has_builtin)
#if __has_builtin(__builtin_amdgcn_global_load_async_to_lds_b128) && \
    __has_builtin(__builtin_amdgcn_s_wait_asynccnt)
#define GFX1250_ASYNC 1
#endif
#endif
#ifndef GFX1250_ASYNC
#define GFX1250_ASYNC 0
#endif

#if GFX1250_ASYNC
typedef __attribute__((address_space(1))) v4i as1_v4i;
typedef __attribute__((address_space(3))) v4i as3_v4i;
static __device__ __forceinline__ void async_copy_b128(const void* g, void* l) {
  // global_load_async_to_lds_b128: per-lane 16B global->LDS, tracked by ASYNCcnt
  __builtin_amdgcn_global_load_async_to_lds_b128((as1_v4i*)g, (as3_v4i*)l, 0, 0);
}
template <int N>
static __device__ __forceinline__ void async_wait() {
  __builtin_amdgcn_s_wait_asynccnt(N);
}
#else
template <int N>
static __device__ __forceinline__ void async_wait() {}
#endif

static __device__ __forceinline__ v8f vzero8() {
  v8f z;
#pragma unroll
  for (int i = 0; i < 8; ++i) z[i] = 0.0f;
  return z;
}

// ---------------------------------------------------------------------------
// f32 -> bf16 conversion (grid-stride)
// ---------------------------------------------------------------------------
__global__ __launch_bounds__(256) void cvt_f32_bf16(const float* __restrict__ src,
                                                    __bf16* __restrict__ dst, int n) {
  int i = blockIdx.x * blockDim.x + threadIdx.x;
  int stride = gridDim.x * blockDim.x;
  for (; i < n; i += stride) dst[i] = (__bf16)src[i];
}

// ---------------------------------------------------------------------------
// Tiled bf16 GEMM:  C[M,N] = (A[M,K] @ B + bias) * scale
//   TRANSB=true : B given as W[N,K] row-major (computes A @ W^T)
//   TRANSB=false: B given as B[K,N] row-major
// Block = 256 threads (8 waves), tile BM x BN, K-step 32, double-buffered LDS.
// Wave grid 4(m) x 2(n); each wave owns 32 x (BN/2) of C as 16x16 WMMA tiles.
// ---------------------------------------------------------------------------
template <int BM, int BN, bool TRANSB, bool OUT_F32>
__global__ __launch_bounds__(256) void gemm_bf16_kernel(
    const __bf16* __restrict__ A, long long strideA,
    const __bf16* __restrict__ Bm, long long strideB,
    const float* __restrict__ bias, float scale,
    void* __restrict__ Cp, long long strideC,
    int M, int Nn, int K) {
  constexpr int KT = 32;
  constexpr int LD = KT + 16;  // 96B row stride: 32B-aligned, spreads banks
  __shared__ __align__(32) __bf16 As[2][BM][LD];
  __shared__ __align__(32) __bf16 BsT[2][BN][LD];  // stored as [n][k]

  // async instructions issued per wave per tile (A chunks + B chunks)
  constexpr int AI = (BM * 4) / 256 + (TRANSB ? (BN * 4) / 256 : 0);

  const int bz = blockIdx.z;
  A  += (long long)bz * strideA;
  Bm += (long long)bz * strideB;

  const int m0 = blockIdx.x * BM;
  const int n0 = blockIdx.y * BN;
  const int t  = threadIdx.x;
  const int w  = t >> 5;
  const int l  = t & 31;
  const int lh = l >> 4;   // which 16-lane half
  const int ll = l & 15;

  constexpr int SM = 2;        // 32 rows per wave
  constexpr int SN = BN / 32;  // 4 (BN=128) or 2 (BN=64)
  const int wmOff = (w & 3) * 32;
  const int wnOff = (w >> 2) * (BN / 2);

  v8f acc[SM][SN];
#pragma unroll
  for (int i = 0; i < SM; ++i)
#pragma unroll
    for (int j = 0; j < SN; ++j) acc[i][j] = vzero8();

  // ---- stagers ----
  auto stageA = [&](int bsel, int k0) {
    for (int c = t; c < BM * 4; c += 256) {
      int row = c >> 2, seg = c & 3;
      const __bf16* src = A + (long long)(m0 + row) * K + k0 + seg * 8;
#if GFX1250_ASYNC
      async_copy_b128(src, &As[bsel][row][seg * 8]);
#else
      *reinterpret_cast<uint4*>(&As[bsel][row][seg * 8]) =
          *reinterpret_cast<const uint4*>(src);
#endif
    }
  };
  auto stageB = [&](int bsel, int k0) {
    if (TRANSB) {
      for (int c = t; c < BN * 4; c += 256) {
        int row = c >> 2, seg = c & 3;
        const __bf16* src = Bm + (long long)(n0 + row) * K + k0 + seg * 8;
#if GFX1250_ASYNC
        async_copy_b128(src, &BsT[bsel][row][seg * 8]);
#else
        *reinterpret_cast<uint4*>(&BsT[bsel][row][seg * 8]) =
            *reinterpret_cast<const uint4*>(src);
#endif
      }
    } else {
      // transpose 2B elements: must go through VGPRs + ds stores
      constexpr int CH = BN / 8;  // 16B chunks along n per k-row
      for (int c = t; c < KT * CH; c += 256) {
        int kk = c / CH;
        int ns = (c % CH) * 8;
        uint4 vv = *reinterpret_cast<const uint4*>(
            Bm + (long long)(k0 + kk) * Nn + n0 + ns);
        const __bf16* pp = reinterpret_cast<const __bf16*>(&vv);
#pragma unroll
        for (int u = 0; u < 8; ++u) BsT[bsel][ns + u][kk] = pp[u];
      }
    }
  };

  // ---- double-buffered main loop ----
  stageA(0, 0);
  stageB(0, 0);
  int buf = 0;
  for (int k0 = 0; k0 < K; k0 += KT) {
    if (k0 + KT < K) {
      stageA(buf ^ 1, k0 + KT);
      stageB(buf ^ 1, k0 + KT);
      async_wait<AI>();  // current tile ready; next tile still in flight
    } else {
      async_wait<0>();
    }
    __syncthreads();  // + compiler-inserted dscnt/loadcnt waits for sync path

    v16bf af[SM], bfr[SN];
#pragma unroll
    for (int i = 0; i < SM; ++i)
      af[i] = *reinterpret_cast<const v16bf*>(&As[buf][wmOff + i * 16 + ll][lh * 16]);
#pragma unroll
    for (int j = 0; j < SN; ++j)
      bfr[j] = *reinterpret_cast<const v16bf*>(&BsT[buf][wnOff + j * 16 + ll][lh * 16]);
#pragma unroll
    for (int i = 0; i < SM; ++i)
#pragma unroll
      for (int j = 0; j < SN; ++j) acc[i][j] = WMMA_BF16(af[i], bfr[j], acc[i][j]);

    __syncthreads();  // protect buf from being restaged while others read it
    buf ^= 1;
  }

  // ---- epilogue: C-layout VGPR g -> rows (g, g+8), lanes 0..15 = cols ----
#pragma unroll
  for (int i = 0; i < SM; ++i) {
#pragma unroll
    for (int j = 0; j < SN; ++j) {
      const int col = n0 + wnOff + j * 16 + ll;
      const float bv = bias ? bias[col] : 0.0f;
#pragma unroll
      for (int g = 0; g < 8; ++g) {
        const int row = m0 + wmOff + i * 16 + g + lh * 8;
        const float val = (acc[i][j][g] + bv) * scale;
        if constexpr (OUT_F32) {
          ((float*)Cp)[(long long)bz * strideC + (long long)row * Nn + col] = val;
        } else {
          ((__bf16*)Cp)[(long long)bz * strideC + (long long)row * Nn + col] = (__bf16)val;
        }
      }
    }
  }
}

// ---------------------------------------------------------------------------
// Fused Linformer flash attention: per (b,h), ctx = softmax(Q Kc^T) Vc.
// Q pre-scaled (d^-0.5 folded into q-projection). P=512 streamed in 64-row
// chunks; K staged via async-to-LDS, V transposed via ds stores.
// Block = 256 threads (8 waves); each wave owns 16 query rows.
// ---------------------------------------------------------------------------
__global__ __launch_bounds__(256) void linformer_attn_kernel(
    const __bf16* __restrict__ qp,   // [B,N,E]
    const __bf16* __restrict__ kp,   // [B,P,E]
    const __bf16* __restrict__ vp,   // [B,P,E]
    __bf16* __restrict__ ctx) {      // [B,N,E]
  constexpr int Pn = 512, D = 64, E = 1024, Nseq = 4096, PC = 64;
  constexpr int LD = 80;  // 160B row stride: 32B-aligned, reduces bank conflicts
  __shared__ __align__(32) __bf16 Kc[PC][LD];      // [p][d]
  __shared__ __align__(32) __bf16 Vt[D][LD];       // [d][p]  (transposed)
  __shared__ __align__(32) __bf16 Pl[8][16][LD];   // per-wave prob tile [m][p]

  const int b = blockIdx.z, h = blockIdx.y;
  const int t = threadIdx.x, w = t >> 5, l = t & 31, lh = l >> 4, ll = l & 15;
  const int m0 = blockIdx.x * 128 + w * 16;  // this wave's query-row base

  // Q A-fragments (16 rows x 64d = 2 fragments of K=32), resident all kernel
  v16bf qf[2];
#pragma unroll
  for (int kk = 0; kk < 2; ++kk)
    qf[kk] = *reinterpret_cast<const v16bf*>(
        qp + ((long long)(b * Nseq + m0 + ll)) * E + h * D + kk * 32 + lh * 16);

  v8f cacc[4];
#pragma unroll
  for (int dt = 0; dt < 4; ++dt) cacc[dt] = vzero8();
  float mrow[8], lrow[8];
#pragma unroll
  for (int g = 0; g < 8; ++g) { mrow[g] = -3.0e38f; lrow[g] = 0.0f; }

  for (int p0 = 0; p0 < Pn; p0 += PC) {
    __syncthreads();  // previous chunk's compute done before restaging
    // stage K chunk [PC][D] (async DMA to LDS)
    for (int c = t; c < PC * 8; c += 256) {
      int p = c >> 3, seg = c & 7;
      const __bf16* src = kp + ((long long)(b * Pn + p0 + p)) * E + h * D + seg * 8;
#if GFX1250_ASYNC
      async_copy_b128(src, &Kc[p][seg * 8]);
#else
      *reinterpret_cast<uint4*>(&Kc[p][seg * 8]) =
          *reinterpret_cast<const uint4*>(src);
#endif
    }
    // stage V chunk transposed: Vt[d][p]
    for (int c = t; c < PC * 8; c += 256) {
      int p = c >> 3, ds = (c & 7) * 8;
      uint4 vv = *reinterpret_cast<const uint4*>(
          vp + ((long long)(b * Pn + p0 + p)) * E + h * D + ds);
      const __bf16* pp = reinterpret_cast<const __bf16*>(&vv);
#pragma unroll
      for (int u = 0; u < 8; ++u) Vt[ds + u][p] = pp[u];
    }
    async_wait<0>();
    __syncthreads();

    // ---- scores S[16][64] = Q @ Kc^T : 4 n-tiles x 2 k-steps of WMMA ----
    v8f s[4];
#pragma unroll
    for (int nt = 0; nt < 4; ++nt) {
      s[nt] = vzero8();
#pragma unroll
      for (int kk = 0; kk < 2; ++kk) {
        v16bf kf = *reinterpret_cast<const v16bf*>(&Kc[nt * 16 + ll][kk * 32 + lh * 16]);
        s[nt] = WMMA_BF16(qf[kk], kf, s[nt]);
      }
    }

    // ---- online softmax (row lives across the 16 lanes of one half) ----
    constexpr float L2E = 1.44269504088896f;
#pragma unroll
    for (int g = 0; g < 8; ++g) {
      float mx = s[0][g];
#pragma unroll
      for (int nt = 1; nt < 4; ++nt) mx = fmaxf(mx, s[nt][g]);
#pragma unroll
      for (int off = 8; off >= 1; off >>= 1) mx = fmaxf(mx, __shfl_xor(mx, off, 16));
      const float nm = fmaxf(mrow[g], mx);
      const float r = exp2f((mrow[g] - nm) * L2E);
      float rs = 0.0f;
#pragma unroll
      for (int nt = 0; nt < 4; ++nt) {
        float pij = exp2f((s[nt][g] - nm) * L2E);
        s[nt][g] = pij;
        rs += pij;
      }
#pragma unroll
      for (int off = 8; off >= 1; off >>= 1) rs += __shfl_xor(rs, off, 16);
      lrow[g] = lrow[g] * r + rs;
      mrow[g] = nm;
#pragma unroll
      for (int dt = 0; dt < 4; ++dt) cacc[dt][g] *= r;
    }

    // ---- P tile -> LDS (C layout) -> reload as A-fragments ----
#pragma unroll
    for (int nt = 0; nt < 4; ++nt)
#pragma unroll
      for (int g = 0; g < 8; ++g)
        Pl[w][g + lh * 8][nt * 16 + ll] = (__bf16)s[nt][g];
    // wave-local LDS RAW: compiler inserts ds-count waits on the dependent loads
#pragma unroll
    for (int kk = 0; kk < 2; ++kk) {
      v16bf pf = *reinterpret_cast<const v16bf*>(&Pl[w][ll][kk * 32 + lh * 16]);
#pragma unroll
      for (int dt = 0; dt < 4; ++dt) {
        v16bf vf = *reinterpret_cast<const v16bf*>(&Vt[dt * 16 + ll][kk * 32 + lh * 16]);
        cacc[dt] = WMMA_BF16(pf, vf, cacc[dt]);
      }
    }
  }

  // ---- normalize + store ctx ----
#pragma unroll
  for (int dt = 0; dt < 4; ++dt)
#pragma unroll
    for (int g = 0; g < 8; ++g) {
      const int row = m0 + g + lh * 8;
      const int col = h * D + dt * 16 + ll;
      ctx[((long long)(b * Nseq + row)) * E + col] = (__bf16)(cacc[dt][g] / lrow[g]);
    }
}

// ---------------------------------------------------------------------------
// Host-side launch
// ---------------------------------------------------------------------------
extern "C" void kernel_launch(void* const* d_in, const int* in_sizes, int n_in,
                              void* d_out, int out_size, void* d_ws, size_t ws_size,
                              hipStream_t stream) {
  (void)in_sizes; (void)n_in; (void)out_size; (void)ws_size;
  constexpr int B = 4, N = 4096, E = 1024, H = 16, P = 512, HD = 64;

  const float* query = (const float*)d_in[0];
  const float* Wq    = (const float*)d_in[1];
  const float* bq    = (const float*)d_in[2];
  const float* Wk    = (const float*)d_in[3];
  const float* bk    = (const float*)d_in[4];
  const float* Wv    = (const float*)d_in[5];
  const float* bv    = (const float*)d_in[6];
  const float* Ek    = (const float*)d_in[7];
  const float* Fv    = (const float*)d_in[8];
  const float* Wo    = (const float*)d_in[9];
  const float* bo    = (const float*)d_in[10];
  const float* Wout  = (const float*)d_in[11];
  const float* bout  = (const float*)d_in[12];

  // ---- workspace carve-up (~101 MB), 256B aligned ----
  char* ws = (char*)d_ws;
  size_t off = 0;
  auto carve = [&](size_t bytes) -> char* {
    char* p = ws + off;
    off += (bytes + 255) & ~(size_t)255;
    return p;
  };
  __bf16* bufA  = (__bf16*)carve((size_t)B * N * E * 2);  // query bf16, later ctx
  __bf16* bufB  = (__bf16*)carve((size_t)B * N * E * 2);  // q-proj bf16, later attn-out
  __bf16* WqB   = (__bf16*)carve((size_t)E * E * 2);
  __bf16* WkB   = (__bf16*)carve((size_t)E * E * 2);
  __bf16* WvB   = (__bf16*)carve((size_t)E * E * 2);
  __bf16* WoB   = (__bf16*)carve((size_t)E * E * 2);
  __bf16* EkB   = (__bf16*)carve((size_t)P * N * 2);
  __bf16* FvB   = (__bf16*)carve((size_t)P * N * 2);
  __bf16* WoutB = (__bf16*)carve((size_t)HD * E * 2);
  __bf16* kcB   = (__bf16*)carve((size_t)B * P * E * 2);
  __bf16* vcB   = (__bf16*)carve((size_t)B * P * E * 2);
  __bf16* kB    = (__bf16*)carve((size_t)B * P * E * 2);
  __bf16* vB    = (__bf16*)carve((size_t)B * P * E * 2);

  auto cvt = [&](const float* s, __bf16* d, int n) {
    int blocks = (n + 1023) / 1024; if (blocks > 4096) blocks = 4096;
    cvt_f32_bf16<<<dim3(blocks), dim3(256), 0, stream>>>(s, d, n);
  };
  cvt(query, bufA, B * N * E);
  cvt(Wq, WqB, E * E);   cvt(Wk, WkB, E * E);
  cvt(Wv, WvB, E * E);   cvt(Wo, WoB, E * E);
  cvt(Ek, EkB, P * N);   cvt(Fv, FvB, P * N);
  cvt(Wout, WoutB, HD * E);

  const float scale = 0.125f;  // d^-0.5, d=64

  // q = (query @ Wq^T + bq) * scale          [B*N, E]
  gemm_bf16_kernel<128, 128, true, false>
      <<<dim3((B * N) / 128, E / 128, 1), 256, 0, stream>>>(
          bufA, 0, WqB, 0, bq, scale, bufB, 0, B * N, E, E);

  // kc[b] = Ek @ query[b]  (A@B form, K = seq)   [P, E] per batch
  gemm_bf16_kernel<128, 128, false, false>
      <<<dim3(P / 128, E / 128, B), 256, 0, stream>>>(
          EkB, 0, bufA, (long long)N * E, nullptr, 1.0f, kcB, (long long)P * E, P, E, N);
  gemm_bf16_kernel<128, 128, false, false>
      <<<dim3(P / 128, E / 128, B), 256, 0, stream>>>(
          FvB, 0, bufA, (long long)N * E, nullptr, 1.0f, vcB, (long long)P * E, P, E, N);

  // k = kc @ Wk^T + bk ;  v = vc @ Wv^T + bv    [B*P, E]
  gemm_bf16_kernel<128, 128, true, false>
      <<<dim3((B * P) / 128, E / 128, 1), 256, 0, stream>>>(
          kcB, 0, WkB, 0, bk, 1.0f, kB, 0, B * P, E, E);
  gemm_bf16_kernel<128, 128, true, false>
      <<<dim3((B * P) / 128, E / 128, 1), 256, 0, stream>>>(
          vcB, 0, WvB, 0, bv, 1.0f, vB, 0, B * P, E, E);

  // fused flash attention -> ctx (reuses bufA; raw query no longer needed)
  linformer_attn_kernel<<<dim3(N / 128, H, B), 256, 0, stream>>>(bufB, kB, vB, bufA);

  // out = ctx @ Wo^T + bo   (reuses bufB; q-proj no longer needed)
  gemm_bf16_kernel<128, 128, true, false>
      <<<dim3((B * N) / 128, E / 128, 1), 256, 0, stream>>>(
          bufA, 0, WoB, 0, bo, 1.0f, bufB, 0, B * N, E, E);

  // final = out @ Wout^T + bout  -> f32 d_out   [B*N, 64]
  gemm_bf16_kernel<128, 64, true, true>
      <<<dim3((B * N) / 128, 1, 1), 256, 0, stream>>>(
          bufB, 0, WoutB, 0, bout, 1.0f, d_out, 0, B * N, HD, E);
}
